// DeepRNN_11708080849334
// MI455X (gfx1250) — matrix operand order repaired and checked by reading, not compiled, not run on hardware
//
#include <hip/hip_runtime.h>
#include <hip/hip_bf16.h>

// Problem constants (match reference)
#define DI 256
#define DH 1024
#define DL 4
#define DC 1000
#define DB 64
#define DT 256

typedef __bf16 bf16_t;
typedef __attribute__((ext_vector_type(16))) __bf16 v16bf;
typedef __attribute__((ext_vector_type(8)))  __bf16 v8bf;
typedef __attribute__((ext_vector_type(8)))  float  v8f;

// ---------------------------------------------------------------------------
// Prep kernels
// ---------------------------------------------------------------------------

// src: f32 row-major [R][Cc]  ->  dst: bf16 row-major [Cc][R]  (transpose+convert)
__global__ void k_transpose_bf16(const float* __restrict__ src,
                                 bf16_t* __restrict__ dst, int R, int Cc) {
  int idx = blockIdx.x * blockDim.x + threadIdx.x;
  int total = R * Cc;
  if (idx >= total) return;
  int r = idx / Cc, c = idx % Cc;
  dst[(size_t)c * R + r] = (bf16_t)src[idx];
}

// x: f32 [B][T][I] -> bf16 time-major [T][B][I]
__global__ void k_convert_x(const float* __restrict__ src, bf16_t* __restrict__ dst) {
  int idx = blockIdx.x * blockDim.x + threadIdx.x;
  if (idx >= DB * DT * DI) return;
  int i = idx % DI;
  int t = (idx / DI) % DT;
  int b = idx / (DI * DT);
  dst[((size_t)t * DB + b) * DI + i] = (bf16_t)src[idx];
}

__global__ void k_zero_bf16(bf16_t* __restrict__ p, int n) {
  int idx = blockIdx.x * blockDim.x + threadIdx.x;
  if (idx < n) p[idx] = (bf16_t)0.0f;
}

// ---------------------------------------------------------------------------
// Layer-step WMMA kernel:  hout = tanh(inp @ WxT' + hprev @ WhT' + bias)
//   inp   : bf16 row-major [DB][Kin]
//   Wxt   : bf16 row-major [DH][Kin]   (transposed weights: Wt[n][k] = W[k][n])
//   hprev : bf16 row-major [DB][DH]
//   Wht   : bf16 row-major [DH][DH]
//   One wave per 16x16 output tile: 256 tiles = 64 blocks x 4 waves.
// ---------------------------------------------------------------------------

__device__ __forceinline__ void gemm_bf16(const bf16_t* __restrict__ Abase,
                                          const bf16_t* __restrict__ Bbase,
                                          int nk, int lh, v8f& acc0, v8f& acc1) {
  for (int k = 0; k < nk; ++k) {
    // A fragment (16x32 bf16): lane holds row lm, 16 contiguous K at 16*lh
    const bf16_t* ap = Abase + 32 * k + 16 * lh;
    v8bf alo = *reinterpret_cast<const v8bf*>(ap);
    v8bf ahi = *reinterpret_cast<const v8bf*>(ap + 8);
    v16bf a = __builtin_shufflevector(alo, ahi,
                                      0, 1, 2, 3, 4, 5, 6, 7,
                                      8, 9, 10, 11, 12, 13, 14, 15);
    // B fragment (32x16 bf16): lane holds column lm, 16 contiguous K at 16*lh
    v16bf b = *reinterpret_cast<const v16bf*>(Bbase + 32 * k + 16 * lh);
    // Stream next weight chunk toward the caches (global_prefetch_b8).
    __builtin_prefetch(Bbase + 32 * (k + 4), 0, 0);
    if (k & 1)
      acc1 = __builtin_amdgcn_wmma_f32_16x16x32_bf16(false, a, false, b,
                                                     (short)0, acc1, false, false);
    else
      acc0 = __builtin_amdgcn_wmma_f32_16x16x32_bf16(false, a, false, b,
                                                     (short)0, acc0, false, false);
  }
}

__global__ __launch_bounds__(128) void k_rnn_layer_step(
    const bf16_t* __restrict__ inp, int Kin,
    const bf16_t* __restrict__ Wxt,
    const bf16_t* __restrict__ hprev,
    const bf16_t* __restrict__ Wht,
    const float*  __restrict__ bias,
    bf16_t* __restrict__ hout) {
  __shared__ bf16_t tile[4][16][17];  // per-wave 16x16 bf16 tile (pad 17)

  const int lane = threadIdx.x & 31;
  const int wave = threadIdx.x >> 5;
  const int t    = blockIdx.x * 4 + wave;  // 0..255
  const int mt   = t & 3;                  // batch tile  (DB/16 = 4)
  const int nt   = t >> 2;                 // column tile (DH/16 = 64)
  const int lm   = lane & 15;
  const int lh   = lane >> 4;

  v8f acc0 = {}, acc1 = {};

  // inp @ Wx
  gemm_bf16(inp   + (size_t)(mt * 16 + lm) * Kin,
            Wxt   + (size_t)(nt * 16 + lm) * Kin, Kin / 32, lh, acc0, acc1);
  // hprev @ Wh
  gemm_bf16(hprev + (size_t)(mt * 16 + lm) * DH,
            Wht   + (size_t)(nt * 16 + lm) * DH, DH / 32, lh, acc0, acc1);

  // bias + tanh.  C/D layout: lane holds column n = lane%16; vgpr i -> row i + 8*(lane/16)
  const float bval = bias[nt * 16 + lm];
#pragma unroll
  for (int i = 0; i < 8; ++i) {
    float v = tanhf(acc0[i] + acc1[i] + bval);
    tile[wave][i + 8 * lh][lm] = (bf16_t)v;  // LDS transpose bounce
  }
  __syncthreads();

  // Read back rows; store row-major bf16 so the next consumer's A loads are plain b128s.
  bf16_t r[8];
#pragma unroll
  for (int i = 0; i < 8; ++i) r[i] = tile[wave][lm][8 * lh + i];
  v8bf out;
#pragma unroll
  for (int i = 0; i < 8; ++i) out[i] = r[i];
  *reinterpret_cast<v8bf*>(hout + (size_t)(mt * 16 + lm) * DH + nt * 16 + 8 * lh) = out;
}

// ---------------------------------------------------------------------------
// Output head: out[b][c] = h[b] . Wo[:,c] + bo[c]   (tiny: 131 MFLOP, plain VALU)
// ---------------------------------------------------------------------------
__global__ void k_head(const bf16_t* __restrict__ h, const float* __restrict__ Wo,
                       const float* __restrict__ bo, float* __restrict__ out) {
  int idx = blockIdx.x * blockDim.x + threadIdx.x;
  if (idx >= DB * DC) return;
  int b = idx / DC, c = idx % DC;
  float s = bo[c];
  const bf16_t* hb = h + (size_t)b * DH;
  for (int k = 0; k < DH; ++k) s += (float)hb[k] * Wo[(size_t)k * DC + c];
  out[idx] = s;
}

// ---------------------------------------------------------------------------
// Host launch
// ---------------------------------------------------------------------------
extern "C" void kernel_launch(void* const* d_in, const int* in_sizes, int n_in,
                              void* d_out, int out_size, void* d_ws, size_t ws_size,
                              hipStream_t stream) {
  (void)in_sizes; (void)n_in; (void)out_size;
  const float* x   = (const float*)d_in[0];  // [B,T,I]
  const float* Wx0 = (const float*)d_in[1];  // [I,H]
  const float* Wx  = (const float*)d_in[2];  // [L-1,H,H]
  const float* Wh  = (const float*)d_in[3];  // [L,H,H]
  const float* bh  = (const float*)d_in[4];  // [L,H]
  const float* Wo  = (const float*)d_in[5];  // [H,C]
  const float* bo  = (const float*)d_in[6];  // [C]

  // Workspace layout (bf16 unless noted)
  size_t o = 0;
  char* w = (char*)d_ws;
  bf16_t* x_bf = (bf16_t*)(w + o); o += (size_t)DT * DB * DI * 2;       //  8 MB
  bf16_t* Wx0t = (bf16_t*)(w + o); o += (size_t)DH * DI * 2;            // .5 MB
  bf16_t* Wxt  = (bf16_t*)(w + o); o += (size_t)(DL - 1) * DH * DH * 2; //  6 MB
  bf16_t* Wht  = (bf16_t*)(w + o); o += (size_t)DL * DH * DH * 2;       //  8 MB
  bf16_t* hbuf = (bf16_t*)(w + o); o += (size_t)2 * DL * DB * DH * 2;   //  1 MB
  if (ws_size < o) return;

  const int TPB = 256;
  // Weight transpose + bf16 convert
  k_transpose_bf16<<<(DI * DH + TPB - 1) / TPB, TPB, 0, stream>>>(Wx0, Wx0t, DI, DH);
  for (int l = 0; l < DL - 1; ++l)
    k_transpose_bf16<<<(DH * DH + TPB - 1) / TPB, TPB, 0, stream>>>(
        Wx + (size_t)l * DH * DH, Wxt + (size_t)l * DH * DH, DH, DH);
  for (int l = 0; l < DL; ++l)
    k_transpose_bf16<<<(DH * DH + TPB - 1) / TPB, TPB, 0, stream>>>(
        Wh + (size_t)l * DH * DH, Wht + (size_t)l * DH * DH, DH, DH);
  // x -> time-major bf16
  k_convert_x<<<(DB * DT * DI + TPB - 1) / TPB, TPB, 0, stream>>>(x, x_bf);
  // h(t=-1) = 0 (ping-pong buffer 0)
  k_zero_bf16<<<(DL * DB * DH + TPB - 1) / TPB, TPB, 0, stream>>>(hbuf, DL * DB * DH);

  // Recurrence: 256 tiles/step -> 64 blocks x 128 threads (4 waves)
  for (int t = 0; t < DT; ++t) {
    bf16_t* hprev = hbuf + (size_t)(t & 1) * DL * DB * DH;
    bf16_t* hcur  = hbuf + (size_t)((t + 1) & 1) * DL * DB * DH;
    for (int l = 0; l < DL; ++l) {
      const bf16_t* inp = (l == 0) ? (x_bf + (size_t)t * DB * DI)
                                   : (hcur + (size_t)(l - 1) * DB * DH);
      int Kin = (l == 0) ? DI : DH;
      const bf16_t* Wxt_l = (l == 0) ? Wx0t : (Wxt + (size_t)(l - 1) * DH * DH);
      k_rnn_layer_step<<<64, 128, 0, stream>>>(
          inp, Kin, Wxt_l,
          hprev + (size_t)l * DB * DH,
          Wht + (size_t)l * DH * DH,
          bh + (size_t)l * DH,
          hcur + (size_t)l * DB * DH);
    }
  }

  // Final hidden state lives in buffer ((T)&1)==0, layer L-1
  const bf16_t* hfin = hbuf + (size_t)(DL - 1) * DB * DH;
  k_head<<<(DB * DC + TPB - 1) / TPB, TPB, 0, stream>>>(hfin, Wo, bo, (float*)d_out);
}
